// MultiheadAttention_50044958933557
// MI455X (gfx1250) — compile-verified, matrix operand order checked by reading, not run on hardware
//
#include <hip/hip_runtime.h>
#include <hip/hip_bf16.h>

// ---------------------------------------------------------------------------
// MI455X (gfx1250) multihead attention, bf16 WMMA path.
//   B=8, H=4, T=2048, E=256, D=64.  D = A*B+C via v_wmma_f32_16x16x32_bf16.
// ~43 GFLOP math vs ~1 GB HBM traffic (avg_weights output alone is 134 MB):
// bf16 16x16x32 WMMA (8x the K/instr of fp32 WMMA) keeps this memory-bound.
// f32->bf16 conversion hoisted to one streaming pre-pass; GEMM inner loops
// are double-buffered b128 loads + WMMA so loads overlap the matrix pipe
// (round-1 version serialized every WMMA behind s_wait_loadcnt 0).
// ---------------------------------------------------------------------------

typedef __bf16 bf16x16 __attribute__((ext_vector_type(16)));
typedef __bf16 bf16x2  __attribute__((ext_vector_type(2)));
typedef float  f32x8   __attribute__((ext_vector_type(8)));
typedef float  f32x2   __attribute__((ext_vector_type(2)));

union FragBF { bf16x16 v; unsigned int u[8]; };

__device__ __forceinline__ unsigned int bfpack(float a, float b) {
  f32x2 f = {a, b};
  union { bf16x2 h; unsigned int u; } c;
  c.h = __builtin_convertvector(f, bf16x2);   // -> v_cvt_pk_bf16_f32
  return c.u;
}
__device__ __forceinline__ unsigned short bf1(float a) {
  union { __bf16 h; unsigned short u; } c;
  c.h = (__bf16)a;
  return c.u;
}
__device__ __forceinline__ f32x8 zero8() {
  f32x8 z = {0.f, 0.f, 0.f, 0.f, 0.f, 0.f, 0.f, 0.f};
  return z;
}
__device__ __forceinline__ f32x8 wmma_bf16(bf16x16 a, bf16x16 b, f32x8 c) {
  return __builtin_amdgcn_wmma_f32_16x16x32_bf16(false, a, false, b,
                                                 (short)0, c, false, false);
}

// A-frag (16xK=32, M = lane&15) or B-frag (Kx16 from a row-major [N][K]
// matrix, i.e. a transposed operand): per ISA 7.12.2, half-wave 0 holds
// K 0..7 & 16..23, half-wave 1 holds K 8..15 & 24..31 -> two b128 loads.
__device__ __forceinline__ bf16x16 frag_u16(const unsigned short* p, int hl) {
  FragBF f;
  uint4 a = *(const uint4*)(p + hl * 8);
  uint4 b = *(const uint4*)(p + 16 + hl * 8);
  f.u[0] = a.x; f.u[1] = a.y; f.u[2] = a.z; f.u[3] = a.w;
  f.u[4] = b.x; f.u[5] = b.y; f.u[6] = b.z; f.u[7] = b.w;
  return f.v;
}

#define B_ 8
#define H_ 4
#define T_ 2048
#define E_ 256
#define D_ 64

// ---------------------------------------------------------------------------
// Kernel 0: streaming f32 -> bf16 convert (query, Wqkv, Wo), b128 in/b64 out.
// ---------------------------------------------------------------------------
__global__ __launch_bounds__(256) void cvt_kernel(
    const float* __restrict__ src, unsigned short* __restrict__ dst, int n4) {
  int i = blockIdx.x * blockDim.x + threadIdx.x;
  if (i < n4) {
    float4 f = ((const float4*)src)[i];
    ((uint2*)dst)[i] = make_uint2(bfpack(f.x, f.y), bfpack(f.z, f.w));
  }
}

// ---------------------------------------------------------------------------
// Kernel 1: qkv = x @ Wqkv^T + b ; scatter into q[B,H,T,D], k[B,H,T,D],
// vT[B,H,D,T] (bf16). q pre-scaled by D^-1/2. 4 waves/block, wave = 16 rows.
// Inner loop: double-buffered B-fragments so b128 loads overlap WMMA.
// ---------------------------------------------------------------------------
__global__ __launch_bounds__(128) void qkv_kernel(
    const unsigned short* __restrict__ xb, const unsigned short* __restrict__ wb,
    const float* __restrict__ bias, unsigned short* __restrict__ qw,
    unsigned short* __restrict__ kw, unsigned short* __restrict__ vtw) {
  const int lane = threadIdx.x & 31;
  const int wave = threadIdx.x >> 5;
  const int hl = lane >> 4;
  const int lr = lane & 15;
  const int row0 = (blockIdx.x * 4 + wave) * 16;  // [0, 16384)

  bf16x16 afr[8];
  const unsigned short* xr = xb + (size_t)(row0 + lr) * E_;
#pragma unroll
  for (int ks = 0; ks < 8; ++ks) afr[ks] = frag_u16(xr + ks * 32, hl);

  const int b = row0 >> 11;
  const int t0 = (row0 & (T_ - 1)) + hl * 8;

  for (int nc = 0; nc < 12; ++nc) {  // 12 * 64 = 768 output cols
    const unsigned short* wrow = wb + (size_t)(nc * 64 + lr) * E_;
    f32x8 c[4] = {zero8(), zero8(), zero8(), zero8()};
    bf16x16 bb[2][4];
#pragma unroll
    for (int nt = 0; nt < 4; ++nt)
      bb[0][nt] = frag_u16(wrow + (size_t)nt * 16 * E_, hl);
#pragma unroll
    for (int ks = 0; ks < 8; ++ks) {
      const int cur = ks & 1, nxt = cur ^ 1;
      if (ks < 7) {  // issue next K-step's loads before consuming this one
#pragma unroll
        for (int nt = 0; nt < 4; ++nt)
          bb[nxt][nt] =
              frag_u16(wrow + (size_t)nt * 16 * E_ + (ks + 1) * 32, hl);
      }
#pragma unroll
      for (int nt = 0; nt < 4; ++nt)
        c[nt] = wmma_bf16(afr[ks], bb[cur][nt], c[nt]);
    }
#pragma unroll
    for (int nt = 0; nt < 4; ++nt) {
      const int f = nc * 64 + nt * 16 + lr;
      const float bv = bias[f];
      const int sect = f >> 8;  // 0=q 1=k 2=v
      const int e = f & (E_ - 1);
      const int hh = e >> 6;
      const int dd = e & (D_ - 1);
      if (sect == 0) {
#pragma unroll
        for (int i = 0; i < 8; ++i)
          qw[((size_t)((b * H_ + hh) * T_ + t0 + i)) * D_ + dd] =
              bf1((c[nt][i] + bv) * 0.125f);  // scaling = 64^-0.5
      } else if (sect == 1) {
#pragma unroll
        for (int i = 0; i < 8; ++i)
          kw[((size_t)((b * H_ + hh) * T_ + t0 + i)) * D_ + dd] =
              bf1(c[nt][i] + bv);
      } else {  // v stored transposed [B,H,D,T]: 8 contiguous t -> one b128
        *(uint4*)(vtw + ((size_t)(b * H_ + hh) * D_ + dd) * T_ + t0) =
            make_uint4(bfpack(c[nt][0] + bv, c[nt][1] + bv),
                       bfpack(c[nt][2] + bv, c[nt][3] + bv),
                       bfpack(c[nt][4] + bv, c[nt][5] + bv),
                       bfpack(c[nt][6] + bv, c[nt][7] + bv));
      }
    }
  }
}

// ---------------------------------------------------------------------------
// Kernel 2: causal attention. One wave = 16 query rows, loops all 4 heads.
// Pass1: streaming row max / sum-exp.  Pass2: recompute S, emit probs to
// avg_weights, transpose P through LDS, accumulate O = P@V (4 WMMA / 32 keys).
// ---------------------------------------------------------------------------
__global__ __launch_bounds__(128) void attn_kernel(
    const unsigned short* __restrict__ qw, const unsigned short* __restrict__ kw,
    const unsigned short* __restrict__ vtw, unsigned short* __restrict__ ctx,
    float* __restrict__ avg) {
  __shared__ __align__(16) unsigned short pshare[4][16 * 32];
  const int lane = threadIdx.x & 31;
  const int wave = threadIdx.x >> 5;
  const int hl = lane >> 4;
  const int lr = lane & 15;
  const int tile = blockIdx.x * 4 + wave;  // [0, 1024)
  const int b = tile >> 7;
  const int r0 = (tile & 127) * 16;        // query row start in T
  unsigned short* ps = pshare[wave];

  for (int h = 0; h < H_; ++h) {
    const unsigned short* qb = qw + (size_t)(b * H_ + h) * T_ * D_;
    const unsigned short* kb = kw + (size_t)(b * H_ + h) * T_ * D_;
    const unsigned short* vb = vtw + (size_t)(b * H_ + h) * D_ * T_;
    bf16x16 aq0 = frag_u16(qb + (size_t)(r0 + lr) * D_, hl);
    bf16x16 aq1 = frag_u16(qb + (size_t)(r0 + lr) * D_ + 32, hl);

    float mr[8], sr[8];
#pragma unroll
    for (int i = 0; i < 8; ++i) { mr[i] = -1e30f; sr[i] = 0.f; }

    // ---- pass 1: online max / sum-exp over causal key tiles ----
    for (int j = 0; j <= r0; j += 16) {
      __builtin_prefetch(kb + (size_t)(j + 16 + lr) * D_, 0, 1);
      bf16x16 bk0 = frag_u16(kb + (size_t)(j + lr) * D_, hl);
      bf16x16 bk1 = frag_u16(kb + (size_t)(j + lr) * D_ + 32, hl);
      f32x8 s = zero8();
      s = wmma_bf16(aq0, bk0, s);
      s = wmma_bf16(aq1, bk1, s);
      const bool diag = (j == r0);
#pragma unroll
      for (int i = 0; i < 8; ++i) {
        float sv = s[i];
        if (diag && (lr > hl * 8 + i)) sv = -1e30f;
        float tm = sv;  // max over the 16 lanes holding this row
        tm = fmaxf(tm, __shfl_xor(tm, 1, 32));
        tm = fmaxf(tm, __shfl_xor(tm, 2, 32));
        tm = fmaxf(tm, __shfl_xor(tm, 4, 32));
        tm = fmaxf(tm, __shfl_xor(tm, 8, 32));
        float mn = fmaxf(mr[i], tm);
        float ev = __expf(sv - mn);
        ev += __shfl_xor(ev, 1, 32);
        ev += __shfl_xor(ev, 2, 32);
        ev += __shfl_xor(ev, 4, 32);
        ev += __shfl_xor(ev, 8, 32);
        sr[i] = sr[i] * __expf(mr[i] - mn) + ev;
        mr[i] = mn;
      }
    }
    float rinv[8];
#pragma unroll
    for (int i = 0; i < 8; ++i) rinv[i] = 1.0f / sr[i];

    // ---- pass 2: probs -> avg_weights, P@V ----
    f32x8 o[4] = {zero8(), zero8(), zero8(), zero8()};
    for (int j0 = 0; j0 <= r0; j0 += 32) {
      const bool has1 = (j0 + 16 <= r0);
      // hoist all K and V fragment loads: in flight during softmax VALU work
      bf16x16 k00 = frag_u16(kb + (size_t)(j0 + lr) * D_, hl);
      bf16x16 k01 = frag_u16(kb + (size_t)(j0 + lr) * D_ + 32, hl);
      bf16x16 k10 = k00, k11 = k01;
      if (has1) {
        k10 = frag_u16(kb + (size_t)(j0 + 16 + lr) * D_, hl);
        k11 = frag_u16(kb + (size_t)(j0 + 16 + lr) * D_ + 32, hl);
      }
      bf16x16 vf[4];
#pragma unroll
      for (int d0 = 0; d0 < 4; ++d0)
        vf[d0] = frag_u16(vb + (size_t)(d0 * 16 + lr) * T_ + j0, hl);

#pragma unroll
      for (int sub = 0; sub < 2; ++sub) {
        const int jt = j0 + sub * 16;
        if (jt <= r0) {
          f32x8 s = zero8();
          s = wmma_bf16(aq0, sub ? k10 : k00, s);
          s = wmma_bf16(aq1, sub ? k11 : k01, s);
          const bool diag = (jt == r0);
#pragma unroll
          for (int i = 0; i < 8; ++i) {
            float sv = s[i];
            if (diag && (lr > hl * 8 + i)) sv = -1e30f;
            float p = __expf(sv - mr[i]) * rinv[i];
            size_t ai = ((size_t)b * T_ + (r0 + hl * 8 + i)) * T_ + jt + lr;
            if (h == 0) avg[ai] = p * 0.25f;   // 1/H
            else        avg[ai] += p * 0.25f;  // wave owns rows: no race
            ps[(hl * 8 + i) * 32 + sub * 16 + lr] = bf1(p);
          }
        } else {  // odd tile count: zero the upper half of the P tile
#pragma unroll
          for (int i = 0; i < 8; ++i) ps[(hl * 8 + i) * 32 + 16 + lr] = 0;
        }
      }
      asm volatile("s_wait_dscnt 0" ::: "memory");  // LDS transpose visible
      bf16x16 pf = frag_u16(ps + lr * 32, hl);      // P as 16x32 A-operand
#pragma unroll
      for (int d0 = 0; d0 < 4; ++d0) o[d0] = wmma_bf16(pf, vf[d0], o[d0]);
    }
    // context out, bf16 [B*T, E]
#pragma unroll
    for (int d0 = 0; d0 < 4; ++d0)
#pragma unroll
      for (int i = 0; i < 8; ++i)
        ctx[((size_t)b * T_ + r0 + hl * 8 + i) * E_ + h * D_ + d0 * 16 + lr] =
            bf1(o[d0][i]);
  }
}

// ---------------------------------------------------------------------------
// Kernel 3: out = ctx @ Wo^T + bo  (f32 result straight into d_out)
// ---------------------------------------------------------------------------
__global__ __launch_bounds__(128) void oproj_kernel(
    const unsigned short* __restrict__ ctx, const unsigned short* __restrict__ wb,
    const float* __restrict__ bias, float* __restrict__ out) {
  const int lane = threadIdx.x & 31;
  const int wave = threadIdx.x >> 5;
  const int hl = lane >> 4;
  const int lr = lane & 15;
  const int row0 = (blockIdx.x * 4 + wave) * 16;

  bf16x16 afr[8];
  const unsigned short* cr = ctx + (size_t)(row0 + lr) * E_;
#pragma unroll
  for (int ks = 0; ks < 8; ++ks) afr[ks] = frag_u16(cr + ks * 32, hl);

  for (int nc = 0; nc < 4; ++nc) {
    const unsigned short* wrow = wb + (size_t)(nc * 64 + lr) * E_;
    f32x8 c[4] = {zero8(), zero8(), zero8(), zero8()};
    bf16x16 bb[2][4];
#pragma unroll
    for (int nt = 0; nt < 4; ++nt)
      bb[0][nt] = frag_u16(wrow + (size_t)nt * 16 * E_, hl);
#pragma unroll
    for (int ks = 0; ks < 8; ++ks) {
      const int cur = ks & 1, nxt = cur ^ 1;
      if (ks < 7) {
#pragma unroll
        for (int nt = 0; nt < 4; ++nt)
          bb[nxt][nt] =
              frag_u16(wrow + (size_t)nt * 16 * E_ + (ks + 1) * 32, hl);
      }
#pragma unroll
      for (int nt = 0; nt < 4; ++nt)
        c[nt] = wmma_bf16(afr[ks], bb[cur][nt], c[nt]);
    }
#pragma unroll
    for (int nt = 0; nt < 4; ++nt) {
      const int f = nc * 64 + nt * 16 + lr;
      const float bv = bias[f];
#pragma unroll
      for (int i = 0; i < 8; ++i)
        out[(size_t)(row0 + hl * 8 + i) * E_ + f] = c[nt][i] + bv;
    }
  }
}

// ---------------------------------------------------------------------------
extern "C" void kernel_launch(void* const* d_in, const int* in_sizes, int n_in,
                              void* d_out, int out_size, void* d_ws,
                              size_t ws_size, hipStream_t stream) {
  const float* query = (const float*)d_in[0];  // (2,4,2048,256)
  const float* wqkv  = (const float*)d_in[1];  // (768,256)
  const float* bqkv  = (const float*)d_in[2];  // (768,)
  const float* wo    = (const float*)d_in[3];  // (256,256)
  const float* bo    = (const float*)d_in[4];  // (256,)

  float* out = (float*)d_out;               // attn: 8*2048*256 f32
  float* avg = out + (size_t)B_ * T_ * E_;  // avg:  8*2048*2048 f32

  const size_t nqkv = (size_t)B_ * H_ * T_ * D_;  // 4,194,304 elems
  const size_t nx = (size_t)B_ * T_ * E_;         // 4,194,304 elems
  const size_t nwq = (size_t)3 * E_ * E_;         // 196,608 elems
  const size_t nwo = (size_t)E_ * E_;             // 65,536 elems
  unsigned short* qws = (unsigned short*)d_ws;
  unsigned short* kws = qws + nqkv;
  unsigned short* vtw = kws + nqkv;
  unsigned short* ctx = vtw + nqkv;
  unsigned short* xb  = ctx + nqkv;
  unsigned short* wqb = xb + nx;
  unsigned short* wob = wqb + nwq;  // total ~42.5 MB of d_ws

  // zero the (never-visited) strictly-upper-triangular part of avg_weights
  hipMemsetAsync(avg, 0, (size_t)B_ * T_ * T_ * sizeof(float), stream);

  cvt_kernel<<<(int)(nx / 4 / 256), 256, 0, stream>>>(query, xb, (int)(nx / 4));
  cvt_kernel<<<(int)(nwq / 4 / 256), 256, 0, stream>>>(wqkv, wqb, (int)(nwq / 4));
  cvt_kernel<<<(int)(nwo / 4 / 256), 256, 0, stream>>>(wo, wob, (int)(nwo / 4));

  qkv_kernel<<<256, 128, 0, stream>>>(xb, wqb, bqkv, qws, kws, vtw);
  attn_kernel<<<256, 128, 0, stream>>>(qws, kws, vtw, ctx, avg);
  oproj_kernel<<<256, 128, 0, stream>>>(ctx, wob, bo, out);
}
